// poMACEExp1NoiseNetwork_32409823215776
// MI455X (gfx1250) — compile-verified, matrix operand order checked by reading, not run on hardware
//
#include <hip/hip_runtime.h>
#include <hip/hip_bf16.h>
#include <math.h>

// ---------------- problem constants (from reference) ----------------
#define NAGENT   8
#define HIDDEN   256
#define SDIM     512
#define EPSZ     256
#define BSZ      512
#define TLEN     64
#define IN_DIM   (NAGENT + HIDDEN + SDIM)   // 776
#define M_TOTAL  (BSZ * TLEN)               // 32768

// ---------------- GEMM tiling ----------------
#define BM 128
#define BN 128
#define BK 32
#define LDSSTRIDE (BK + 8)                  // 40 halves = 80B rows (16B aligned)

typedef __attribute__((ext_vector_type(16))) __bf16 v16bf;
typedef __attribute__((ext_vector_type(8)))  float  v8f;
typedef __attribute__((__vector_size__(16))) int    v4i_t;

#if __has_builtin(__builtin_amdgcn_global_load_async_to_lds_b128)
#define USE_ASYNC_B 1
// global source (addrspace(1)) / LDS destination (addrspace(3)) as v4i*
#define GPTR(p) ((__attribute__((address_space(1))) v4i_t*)(void*)(p))
#define LPTR(p) ((__attribute__((address_space(3))) v4i_t*)(void*)(p))
#else
#define USE_ASYNC_B 0
#endif

// ---------------- helpers ----------------
#if __has_builtin(__builtin_amdgcn_cvt_pk_bf16_f32)
typedef __attribute__((ext_vector_type(2))) __bf16 v2bf;
__device__ __forceinline__ unsigned pk2bf(float x, float y) {
    union { v2bf v; unsigned u; } c;
    c.v = __builtin_amdgcn_cvt_pk_bf16_f32(x, y);
    return c.u;
}
#else
// round-biased truncate + v_perm_b32 pack: 3 VALU per bf16 pair
__device__ __forceinline__ unsigned pk2bf(float x, float y) {
    unsigned ux = __float_as_uint(x) + 0x7FFFu;
    unsigned uy = __float_as_uint(y) + 0x7FFFu;
    // result bytes (LSB..MSB) = {ux[2], ux[3], uy[2], uy[3]}
    return __builtin_amdgcn_perm(uy, ux, 0x07060302u);
}
#endif

__device__ __forceinline__ uint2 threefry2x32(uint2 key, uint2 ctr) {
    const unsigned RA[4] = {13u, 15u, 26u, 6u};
    const unsigned RB[4] = {17u, 29u, 16u, 24u};
    unsigned ks0 = key.x, ks1 = key.y;
    unsigned ks2 = 0x1BD11BDAu ^ ks0 ^ ks1;
    unsigned x0 = ctr.x + ks0;
    unsigned x1 = ctr.y + ks1;
    #pragma unroll
    for (int blk = 0; blk < 5; ++blk) {
        const unsigned* R = (blk & 1) ? RB : RA;
        #pragma unroll
        for (int r = 0; r < 4; ++r) {
            x0 += x1;
            x1 = (x1 << R[r]) | (x1 >> (32u - R[r]));
            x1 ^= x0;
        }
        switch (blk) {
            case 0: x0 += ks1; x1 += ks2 + 1u; break;
            case 1: x0 += ks2; x1 += ks0 + 2u; break;
            case 2: x0 += ks0; x1 += ks1 + 3u; break;
            case 3: x0 += ks1; x1 += ks2 + 4u; break;
            case 4: x0 += ks2; x1 += ks0 + 5u; break;
        }
    }
    return make_uint2(x0, x1);
}

// ---------------- kernel 1: epsilons + ax (per batch b) ----------------
__global__ __launch_bounds__(256)
void prep_eps_ax(const float* __restrict__ xi,
                 const float* __restrict__ agent_id,
                 const float* __restrict__ eps_var,
                 const long long* __restrict__ seeds,
                 const float* __restrict__ W,
                 const float* __restrict__ bias,
                 float* __restrict__ axbuf,
                 float* __restrict__ epsbuf) {
    const int b = blockIdx.x;      // 0..511
    const int e = threadIdx.x;     // 0..255

    __shared__ float xi_sh[NAGENT][HIDDEN];
    __shared__ float aid_sh[NAGENT][NAGENT];
    #pragma unroll
    for (int a = 0; a < NAGENT; ++a)
        xi_sh[a][e] = xi[((size_t)a * BSZ + b) * HIDDEN + e];
    if (e < NAGENT * NAGENT) {
        int a = e >> 3, j = e & 7;
        aid_sh[a][j] = agent_id[((size_t)a * BSZ + b) * NAGENT + j];
    }
    __syncthreads();

    // epsilon: threefry fold_in(key(0), seed) -> per-element gaussian * variance
    unsigned s = (unsigned)(seeds[b] & 0xFFFFFFFFll);
    uint2 fk = threefry2x32(make_uint2(0u, 0u), make_uint2(0u, s));
    uint2 rb = threefry2x32(fk, make_uint2((unsigned)e, 0x9E3779B9u));
    float u1 = fmaxf((float)(rb.x >> 8) * (1.0f / 16777216.0f), 1e-9f);
    float u2 = (float)(rb.y >> 8) * (1.0f / 16777216.0f);
    float z  = sqrtf(-2.0f * logf(u1)) * __cosf(6.283185307179586f * u2);
    epsbuf[(size_t)b * EPSZ + e] = z * eps_var[b];

    // ax[a,b,e] = bias[e] + agent_id[a,b,:8].Wa[e,:] + xi[a,b,:256].Wx[e,:]
    const float* wrow = W + (size_t)e * IN_DIM;   // row e of W (L2-resident)
    #pragma unroll
    for (int a = 0; a < NAGENT; ++a) {
        float acc = bias[e];
        #pragma unroll
        for (int j = 0; j < NAGENT; ++j)
            acc += aid_sh[a][j] * wrow[j];
        for (int h = 0; h < HIDDEN; ++h)
            acc += xi_sh[a][h] * wrow[NAGENT + h];
        axbuf[((size_t)a * BSZ + b) * EPSZ + e] = acc;
    }
}

// ---------------- kernel 2: Ws (f32) -> bf16, row-major [n][k] ----------------
__global__ __launch_bounds__(256)
void prep_wsB(const float* __restrict__ W, unsigned short* __restrict__ wsB) {
    const int n = blockIdx.x;          // 0..255 (output feature)
    const int t = threadIdx.x;         // 0..255
    const int k = t * 2;
    const float* src = W + (size_t)n * IN_DIM + (NAGENT + HIDDEN);
    *(unsigned*)&wsB[(size_t)n * SDIM + k] = pk2bf(src[k], src[k + 1]);
}

// ---------------- kernel 3: WMMA GEMM (st = state.Ws^T) + fused epilogue ----
// out[a, r, c] = relu(st[r,c] + ax[a, r/64, c]) * eps[r/64, c]
__global__ __launch_bounds__(256)
void gemm_fused(const float* __restrict__ state,
                const unsigned short* __restrict__ wsB,
                const float* __restrict__ axbuf,
                const float* __restrict__ epsbuf,
                float* __restrict__ out) {
    __shared__ __align__(16) unsigned short As[BM * LDSSTRIDE];
    __shared__ __align__(16) unsigned short Bs[BN * LDSSTRIDE];

    const int tid  = threadIdx.x;
    const int lane = tid & 31;
    const int wave = tid >> 5;
    const int wm   = wave & 3;     // 4 waves along M (32 rows each)
    const int wn   = wave >> 2;    // 2 waves along N (64 cols each)
    const int rowBase = blockIdx.x * BM;
    const int colBase = blockIdx.y * BN;

    v8f acc[2][4];
    #pragma unroll
    for (int mt = 0; mt < 2; ++mt)
        #pragma unroll
        for (int nt = 0; nt < 4; ++nt)
            acc[mt][nt] = (v8f){0.f, 0.f, 0.f, 0.f, 0.f, 0.f, 0.f, 0.f};

    // staging assignment: 2 threads per row, 16 elems each
    const int ldr = tid >> 1;             // 0..127
    const int ldc = (tid & 1) * 16;       // 0 or 16
    const float*          aSrc = state + (size_t)(rowBase + ldr) * SDIM + ldc;
    const unsigned short* bSrc = wsB   + (size_t)(colBase + ldr) * SDIM + ldc;

    const int hi  = (lane >> 4) & 1;      // K-half select per ISA layout
    const int off = hi * 8;

    for (int kk = 0; kk < SDIM; kk += BK) {
        // ---- stage B: pure bf16 copy -> async DMA to LDS when available ----
#if USE_ASYNC_B
        __builtin_amdgcn_global_load_async_to_lds_b128(
            GPTR(bSrc + kk),
            LPTR(&Bs[ldr * LDSSTRIDE + ldc]),
            0, 0);
        __builtin_amdgcn_global_load_async_to_lds_b128(
            GPTR(bSrc + kk + 8),
            LPTR(&Bs[ldr * LDSSTRIDE + ldc + 8]),
            0, 0);
#else
        uint4 b0 = *(const uint4*)(bSrc + kk);
        uint4 b1 = *(const uint4*)(bSrc + kk + 8);
        *(uint4*)&Bs[ldr * LDSSTRIDE + ldc]     = b0;
        *(uint4*)&Bs[ldr * LDSSTRIDE + ldc + 8] = b1;
#endif

        // ---- stage A (f32 -> bf16 via biased-round + v_perm_b32 pack) ----
        union { unsigned u[8]; uint4 q[2]; } ab;
        #pragma unroll
        for (int v = 0; v < 4; ++v) {
            float4 f = *(const float4*)(aSrc + kk + v * 4);
            ab.u[v * 2 + 0] = pk2bf(f.x, f.y);
            ab.u[v * 2 + 1] = pk2bf(f.z, f.w);
        }
        *(uint4*)&As[ldr * LDSSTRIDE + ldc]     = ab.q[0];
        *(uint4*)&As[ldr * LDSSTRIDE + ldc + 8] = ab.q[1];

        if (kk + BK < SDIM)
            __builtin_prefetch(aSrc + kk + BK, 0, 0);   // global_prefetch_b8

#if USE_ASYNC_B
#if __has_builtin(__builtin_amdgcn_s_wait_asynccnt)
        __builtin_amdgcn_s_wait_asynccnt(0);
#else
        asm volatile("s_wait_asynccnt 0x0" ::: "memory");
#endif
#endif
        __syncthreads();

        // ---- fragments per ISA 16-bit 16x32 layout ----
        union Frag { uint4 u[2]; v16bf v; };
        Frag aF[2], bF[4];
        #pragma unroll
        for (int mt = 0; mt < 2; ++mt) {
            int r = wm * 32 + mt * 16 + (lane & 15);
            aF[mt].u[0] = *(const uint4*)&As[r * LDSSTRIDE + off];
            aF[mt].u[1] = *(const uint4*)&As[r * LDSSTRIDE + off + 16];
        }
        #pragma unroll
        for (int nt = 0; nt < 4; ++nt) {
            int c = wn * 64 + nt * 16 + (lane & 15);
            bF[nt].u[0] = *(const uint4*)&Bs[c * LDSSTRIDE + off];
            bF[nt].u[1] = *(const uint4*)&Bs[c * LDSSTRIDE + off + 16];
        }

        #pragma unroll
        for (int mt = 0; mt < 2; ++mt)
            #pragma unroll
            for (int nt = 0; nt < 4; ++nt)
                acc[mt][nt] = __builtin_amdgcn_wmma_f32_16x16x32_bf16(
                    false, aF[mt].v, false, bF[nt].v,
                    (short)0, acc[mt][nt], false, false);

        __syncthreads();
    }

    // ---- fused epilogue: broadcast st tile to all 8 agents ----
    // C/D layout: VGPR i -> lanes 0-15: M=i, lanes 16-31: M=i+8 ; N = lane%16
    // A 16-row tile is 16-aligned, batch blocks are 64 rows -> one bIdx per tile.
    #pragma unroll
    for (int mt = 0; mt < 2; ++mt) {
        const int rTile = rowBase + wm * 32 + mt * 16;
        const int bIdx  = rTile >> 6;                  // r = b*T + t
        #pragma unroll
        for (int nt = 0; nt < 4; ++nt) {
            const int cGlob = colBase + wn * 64 + nt * 16 + (lane & 15);
            const float epsv = epsbuf[(size_t)bIdx * EPSZ + cGlob];
            float axv[NAGENT];
            #pragma unroll
            for (int a = 0; a < NAGENT; ++a)
                axv[a] = axbuf[((size_t)a * BSZ + bIdx) * EPSZ + cGlob];
            #pragma unroll
            for (int i = 0; i < 8; ++i) {
                const int rGlob = rTile + (hi ? (i + 8) : i);
                const float stv = acc[mt][nt][i];
                float* orow = out + (size_t)rGlob * EPSZ + cGlob;
                #pragma unroll
                for (int a = 0; a < NAGENT; ++a) {
                    float sig = stv + axv[a];
                    sig = sig > 0.f ? sig : 0.f;
                    orow[(size_t)a * ((size_t)M_TOTAL * EPSZ)] = sig * epsv;
                }
            }
        }
    }
}

// ---------------- launcher ----------------
extern "C" void kernel_launch(void* const* d_in, const int* in_sizes, int n_in,
                              void* d_out, int out_size, void* d_ws, size_t ws_size,
                              hipStream_t stream) {
    const float*     state    = (const float*)d_in[0];       // (512,64,512)
    const float*     xi       = (const float*)d_in[1];       // (8,512,1,256)
    const float*     agent_id = (const float*)d_in[2];       // (8,512,1,8)
    const float*     eps_var  = (const float*)d_in[3];       // (512,1)
    const long long* seeds    = (const long long*)d_in[4];   // (512,1) int64
    const float*     W        = (const float*)d_in[5];       // (256,776)
    const float*     bias     = (const float*)d_in[6];       // (256,)
    float*           out      = (float*)d_out;               // (8,512,64,256)

    char* ws = (char*)d_ws;
    float* axbuf = (float*)ws;                                            // 4 MB
    float* epsbuf = (float*)(ws + (size_t)NAGENT * BSZ * EPSZ * 4);       // 512 KB
    unsigned short* wsB =
        (unsigned short*)(ws + (size_t)NAGENT * BSZ * EPSZ * 4
                             + (size_t)BSZ * EPSZ * 4);                   // 256 KB

    hipLaunchKernelGGL(prep_eps_ax, dim3(BSZ), dim3(256), 0, stream,
                       xi, agent_id, eps_var, seeds, W, bias, axbuf, epsbuf);
    hipLaunchKernelGGL(prep_wsB, dim3(EPSZ), dim3(256), 0, stream, W, wsB);
    hipLaunchKernelGGL(gemm_fused, dim3(M_TOTAL / BM, 2),
                       dim3(256), 0, stream, state, wsB, axbuf, epsbuf, out);
}
